// GAT_61194694034152
// MI455X (gfx1250) — compile-verified, matrix operand order checked by reading, not run on hardware
//
#include <hip/hip_runtime.h>
#include <hip/hip_bf16.h>
#include <math.h>

// ---------------- types / helpers ----------------
typedef __attribute__((ext_vector_type(16))) _Float16 v16h;
typedef __attribute__((ext_vector_type(8)))  float    v8f;

union U16h { v16h v; uint4 q[2]; };

#define F_IN 256
#define HD   256      // H*D = 4*64
#define NH   4
#define DH   64
#define NC   16
#define ORD_NEG_INF (-2139095041)   // ordf(-inf) == 0x807FFFFF

__device__ __forceinline__ int ordf(float f) {
  int i = __float_as_int(f);
  return (i >= 0) ? i : (i ^ 0x7fffffff);
}
__device__ __forceinline__ float unordf(int i) {
  int j = (i >= 0) ? i : (i ^ 0x7fffffff);
  return __int_as_float(j);
}

// ---------------- fills ----------------
__global__ void k_fill_f32(float* p, float v, int n) {
  int i = blockIdx.x * blockDim.x + threadIdx.x;
  if (i < n) p[i] = v;
}
__global__ void k_fill_i32(int* p, int v, int n) {
  int i = blockIdx.x * blockDim.x + threadIdx.x;
  if (i < n) p[i] = v;
}

// ---------------- f32 -> f16 conversions ----------------
__global__ void k_cvt_f16(const float* __restrict__ src, _Float16* __restrict__ dst, int n) {
  int i = blockIdx.x * blockDim.x + threadIdx.x;
  if (i < n) dst[i] = (_Float16)src[i];
}
// W1 [256 in][256 out] -> W1T f16 [256 out][256 in]
__global__ void k_w1t(const float* __restrict__ W, _Float16* __restrict__ WT) {
  int i = blockIdx.x * blockDim.x + threadIdx.x;   // 256*256
  int o = i >> 8, in = i & 255;
  WT[o * 256 + in] = (_Float16)W[in * 256 + o];
}
// W2 [256 in][16 out] -> W2T f16 [16 out][256 in]
__global__ void k_w2t(const float* __restrict__ W, _Float16* __restrict__ WT) {
  int i = blockIdx.x * blockDim.x + threadIdx.x;   // 16*256
  if (i >= 16 * 256) return;
  int o = i >> 8, in = i & 255;
  WT[o * 256 + in] = (_Float16)W[in * 16 + o];
}

// ---------------- GEMM1: h1[N,256] = x_f16[N,256] @ W1T_f16 ----------------
// one wave -> 16x64 tile (4 wmma accumulators), K=256 in 8 steps of 32
__global__ void k_gemm1(const _Float16* __restrict__ A, const _Float16* __restrict__ B,
                        float* __restrict__ C, int Nn) {
  int wid  = (blockIdx.x * blockDim.x + threadIdx.x) >> 5;
  int lane = threadIdx.x & 31;
  int mt = wid >> 2;          // M tile (16 rows)
  int ng = wid & 3;           // 64-col group
  if (mt * 16 >= Nn) return;
  int lm = lane & 15;
  int kb = (lane < 16) ? 0 : 8;

  const _Float16* arow = A + (size_t)(mt * 16 + lm) * 256 + kb;
  const _Float16* br0  = B + (size_t)(ng * 64 +  0 + lm) * 256 + kb;
  const _Float16* br1  = B + (size_t)(ng * 64 + 16 + lm) * 256 + kb;
  const _Float16* br2  = B + (size_t)(ng * 64 + 32 + lm) * 256 + kb;
  const _Float16* br3  = B + (size_t)(ng * 64 + 48 + lm) * 256 + kb;

  v8f acc0 = {}, acc1 = {}, acc2 = {}, acc3 = {};
  for (int k = 0; k < 256; k += 32) {
    __builtin_prefetch(arow + k + 64, 0, 1);   // global_prefetch_b8 (gfx1250)
    U16h a;
    a.q[0] = *(const uint4*)(arow + k);
    a.q[1] = *(const uint4*)(arow + k + 16);
    U16h b;
    b.q[0] = *(const uint4*)(br0 + k); b.q[1] = *(const uint4*)(br0 + k + 16);
    acc0 = __builtin_amdgcn_wmma_f32_16x16x32_f16(false, a.v, false, b.v, (short)0, acc0, false, false);
    b.q[0] = *(const uint4*)(br1 + k); b.q[1] = *(const uint4*)(br1 + k + 16);
    acc1 = __builtin_amdgcn_wmma_f32_16x16x32_f16(false, a.v, false, b.v, (short)0, acc1, false, false);
    b.q[0] = *(const uint4*)(br2 + k); b.q[1] = *(const uint4*)(br2 + k + 16);
    acc2 = __builtin_amdgcn_wmma_f32_16x16x32_f16(false, a.v, false, b.v, (short)0, acc2, false, false);
    b.q[0] = *(const uint4*)(br3 + k); b.q[1] = *(const uint4*)(br3 + k + 16);
    acc3 = __builtin_amdgcn_wmma_f32_16x16x32_f16(false, a.v, false, b.v, (short)0, acc3, false, false);
  }
  int mhi = (lane >> 4) << 3;   // +8 rows for lanes 16..31
#pragma unroll
  for (int r = 0; r < 8; r++) {
    size_t row = (size_t)(mt * 16 + r + mhi) * 256 + ng * 64 + lm;
    C[row +  0] = acc0[r];
    C[row + 16] = acc1[r];
    C[row + 32] = acc2[r];
    C[row + 48] = acc3[r];
  }
}

// ---------------- GEMM2: g[N,16] = h2_f16[N,256] @ W2T_f16 ----------------
__global__ void k_gemm2(const _Float16* __restrict__ A, const _Float16* __restrict__ B,
                        float* __restrict__ C, int Nn) {
  int wid  = (blockIdx.x * blockDim.x + threadIdx.x) >> 5;
  int lane = threadIdx.x & 31;
  if (wid * 16 >= Nn) return;
  int lm = lane & 15;
  int kb = (lane < 16) ? 0 : 8;
  const _Float16* arow = A + (size_t)(wid * 16 + lm) * 256 + kb;
  const _Float16* brow = B + (size_t)lm * 256 + kb;
  v8f acc = {};
  for (int k = 0; k < 256; k += 32) {
    U16h a, b;
    a.q[0] = *(const uint4*)(arow + k); a.q[1] = *(const uint4*)(arow + k + 16);
    b.q[0] = *(const uint4*)(brow + k); b.q[1] = *(const uint4*)(brow + k + 16);
    acc = __builtin_amdgcn_wmma_f32_16x16x32_f16(false, a.v, false, b.v, (short)0, acc, false, false);
  }
  int mhi = (lane >> 4) << 3;
#pragma unroll
  for (int r = 0; r < 8; r++)
    C[(size_t)(wid * 16 + r + mhi) * 16 + lm] = acc[r];
}

// ---------------- attention logits ----------------
// layer1: one wave per (node, head); two 64-dim dot products
__global__ void k_alpha1(const float* __restrict__ h1, const float* __restrict__ as,
                         const float* __restrict__ ad, float* __restrict__ als,
                         float* __restrict__ ald, int Nn) {
  int wid  = (blockIdx.x * blockDim.x + threadIdx.x) >> 5;
  int lane = threadIdx.x & 31;
  if (wid >= Nn * NH) return;
  int n = wid >> 2, hh = wid & 3;
  const float* hp = h1 + (size_t)n * 256 + hh * 64;
  float sv = 0.f, dv = 0.f;
  for (int d = lane; d < 64; d += 32) {
    float hvv = hp[d];
    sv += hvv * as[hh * 64 + d];
    dv += hvv * ad[hh * 64 + d];
  }
  for (int off = 16; off; off >>= 1) {
    sv += __shfl_xor(sv, off, 32);
    dv += __shfl_xor(dv, off, 32);
  }
  if (lane == 0) { als[wid] = sv; ald[wid] = dv; }
}
// layer2: one thread per node, 16-dim dots
__global__ void k_alpha2(const float* __restrict__ g, const float* __restrict__ as,
                         const float* __restrict__ ad, float* __restrict__ als,
                         float* __restrict__ ald, int Nn) {
  int n = blockIdx.x * blockDim.x + threadIdx.x;
  if (n >= Nn) return;
  float sv = 0.f, dv = 0.f;
#pragma unroll
  for (int c = 0; c < NC; c++) {
    float gv = g[(size_t)n * NC + c];
    sv += gv * as[c]; dv += gv * ad[c];
  }
  als[n] = sv; ald[n] = dv;
}

// ---------------- edge softmax passes, layer 1 (H=4) ----------------
__global__ void k_edge_max1(const int* __restrict__ ei, int E, int Etot,
                            const float* __restrict__ als, const float* __restrict__ ald,
                            int* __restrict__ m) {
  int e = blockIdx.x * blockDim.x + threadIdx.x;
  if (e >= Etot) return;
  int src, dst;
  if (e < E) { src = ei[e]; dst = ei[E + e]; } else { src = dst = e - E; }
#pragma unroll
  for (int hh = 0; hh < NH; hh++) {
    float v = als[src * NH + hh] + ald[dst * NH + hh];
    v = v > 0.f ? v : 0.2f * v;
    atomicMax(m + dst * NH + hh, ordf(v));
  }
}
__global__ void k_edge_sum1(const int* __restrict__ ei, int E, int Etot,
                            const float* __restrict__ als, const float* __restrict__ ald,
                            const int* __restrict__ m, float* __restrict__ s,
                            float* __restrict__ ex) {
  int e = blockIdx.x * blockDim.x + threadIdx.x;
  if (e >= Etot) return;
  int src, dst;
  if (e < E) { src = ei[e]; dst = ei[E + e]; } else { src = dst = e - E; }
#pragma unroll
  for (int hh = 0; hh < NH; hh++) {
    float v = als[src * NH + hh] + ald[dst * NH + hh];
    v = v > 0.f ? v : 0.2f * v;
    float evv = __expf(v - unordf(m[dst * NH + hh]));
    ex[(size_t)e * NH + hh] = evv;
    atomicAdd(s + dst * NH + hh, evv);
  }
}
// one block (256 threads) per edge: coalesced gather of h1[src,:] and scatter-add
__global__ void k_edge_agg1(const int* __restrict__ ei, int E, int Etot,
                            const float* __restrict__ h1, const float* __restrict__ ex,
                            const float* __restrict__ s, float* __restrict__ agg) {
  int e = blockIdx.x;
  int c = threadIdx.x;           // 0..255
  int hh = c >> 6;
  int src, dst;
  if (e < E) { src = ei[e]; dst = ei[E + e]; } else { src = dst = e - E; }
  float att = ex[(size_t)e * NH + hh] / (s[dst * NH + hh] + 1e-16f);
  atomicAdd(agg + (size_t)dst * 256 + c, h1[(size_t)src * 256 + c] * att);
}

// bias + ELU + f16 convert for layer-2 GEMM input
__global__ void k_elu1(const float* __restrict__ agg, const float* __restrict__ bias,
                       _Float16* __restrict__ out, int n) {
  int i = blockIdx.x * blockDim.x + threadIdx.x;
  if (i >= n) return;
  float v = agg[i] + bias[i & 255];
  v = v > 0.f ? v : (__expf(v) - 1.0f);
  out[i] = (_Float16)v;
}

// ---------------- edge softmax passes, layer 2 (H=1, D=16) ----------------
__global__ void k_edge_max2(const int* __restrict__ ei, int E, int Etot,
                            const float* __restrict__ als, const float* __restrict__ ald,
                            int* __restrict__ m) {
  int e = blockIdx.x * blockDim.x + threadIdx.x;
  if (e >= Etot) return;
  int src, dst;
  if (e < E) { src = ei[e]; dst = ei[E + e]; } else { src = dst = e - E; }
  float v = als[src] + ald[dst];
  v = v > 0.f ? v : 0.2f * v;
  atomicMax(m + dst, ordf(v));
}
__global__ void k_edge_sum2(const int* __restrict__ ei, int E, int Etot,
                            const float* __restrict__ als, const float* __restrict__ ald,
                            const int* __restrict__ m, float* __restrict__ s,
                            float* __restrict__ ex) {
  int e = blockIdx.x * blockDim.x + threadIdx.x;
  if (e >= Etot) return;
  int src, dst;
  if (e < E) { src = ei[e]; dst = ei[E + e]; } else { src = dst = e - E; }
  float v = als[src] + ald[dst];
  v = v > 0.f ? v : 0.2f * v;
  float evv = __expf(v - unordf(m[dst]));
  ex[e] = evv;
  atomicAdd(s + dst, evv);
}
__global__ void k_edge_agg2(const int* __restrict__ ei, int E, int Etot,
                            const float* __restrict__ g, const float* __restrict__ ex,
                            const float* __restrict__ s, float* __restrict__ agg) {
  int t = blockIdx.x * blockDim.x + threadIdx.x;
  int e = t >> 4, c = t & 15;
  if (e >= Etot) return;
  int src, dst;
  if (e < E) { src = ei[e]; dst = ei[E + e]; } else { src = dst = e - E; }
  float att = ex[e] / (s[dst] + 1e-16f);
  atomicAdd(agg + (size_t)dst * NC + c, g[(size_t)src * NC + c] * att);
}

// ---------------- final bias + log_softmax over 16 classes ----------------
__global__ void k_logsoftmax(const float* __restrict__ agg, const float* __restrict__ bias,
                             float* __restrict__ out, int Nn) {
  int n = blockIdx.x * blockDim.x + threadIdx.x;
  if (n >= Nn) return;
  float v[NC];
  float mx = -INFINITY;
#pragma unroll
  for (int c = 0; c < NC; c++) { v[c] = agg[(size_t)n * NC + c] + bias[c]; mx = fmaxf(mx, v[c]); }
  float sum = 0.f;
#pragma unroll
  for (int c = 0; c < NC; c++) sum += __expf(v[c] - mx);
  float ls = __logf(sum);
#pragma unroll
  for (int c = 0; c < NC; c++) out[(size_t)n * NC + c] = v[c] - mx - ls;
}

// ---------------- host launcher ----------------
extern "C" void kernel_launch(void* const* d_in, const int* in_sizes, int n_in,
                              void* d_out, int out_size, void* d_ws, size_t ws_size,
                              hipStream_t stream) {
  const float* x      = (const float*)d_in[0];
  const int*   ei     = (const int*)  d_in[1];
  const float* W1     = (const float*)d_in[2];
  const float* a_src1 = (const float*)d_in[3];
  const float* a_dst1 = (const float*)d_in[4];
  const float* b1     = (const float*)d_in[5];
  const float* W2     = (const float*)d_in[6];
  const float* a_src2 = (const float*)d_in[7];
  const float* a_dst2 = (const float*)d_in[8];
  const float* b2     = (const float*)d_in[9];
  float* out = (float*)d_out;

  const int Nn   = in_sizes[0] / F_IN;
  const int E    = in_sizes[1] / 2;
  const int Etot = E + Nn;

  // ---- workspace carve-up (256B aligned) ----
  char* w = (char*)d_ws;
  size_t off = 0;
  auto take = [&](size_t bytes) -> char* {
    char* p = w + off;
    off = (off + bytes + 255) & ~(size_t)255;
    return p;
  };
  float*    h1    = (float*)   take((size_t)Nn * 256 * 4);
  float*    agg1  = (float*)   take((size_t)Nn * 256 * 4);
  _Float16* xh    = (_Float16*)take((size_t)Nn * 256 * 2);
  _Float16* h2h   = (_Float16*)take((size_t)Nn * 256 * 2);
  _Float16* W1T   = (_Float16*)take(256 * 256 * 2);
  _Float16* W2T   = (_Float16*)take(16 * 256 * 2);
  float*    als1  = (float*)   take((size_t)Nn * NH * 4);
  float*    ald1  = (float*)   take((size_t)Nn * NH * 4);
  int*      m1    = (int*)     take((size_t)Nn * NH * 4);
  float*    s1    = (float*)   take((size_t)Nn * NH * 4);
  float*    ex1   = (float*)   take((size_t)Etot * NH * 4);
  float*    g     = (float*)   take((size_t)Nn * NC * 4);
  float*    agg2  = (float*)   take((size_t)Nn * NC * 4);
  float*    als2  = (float*)   take((size_t)Nn * 4);
  float*    ald2  = (float*)   take((size_t)Nn * 4);
  int*      m2    = (int*)     take((size_t)Nn * 4);
  float*    s2    = (float*)   take((size_t)Nn * 4);
  float*    ex2   = (float*)   take((size_t)Etot * 4);
  (void)ws_size; (void)n_in; (void)out_size;

  const int T = 256;
  auto blk = [](long long n, int t) { return (unsigned)((n + t - 1) / t); };

  // f16 conversions
  k_cvt_f16<<<blk((long long)Nn * 256, T), T, 0, stream>>>(x, xh, Nn * 256);
  k_w1t<<<blk(256 * 256, T), T, 0, stream>>>(W1, W1T);
  k_w2t<<<blk(16 * 256, T), T, 0, stream>>>(W2, W2T);

  // GEMM1 (WMMA): h1 = x @ W1
  {
    long long waves = (long long)(Nn / 16) * 4;
    k_gemm1<<<blk(waves * 32, T), T, 0, stream>>>(xh, W1T, h1, Nn);
  }

  // layer-1 attention logits
  k_alpha1<<<blk((long long)Nn * NH * 32, T), T, 0, stream>>>(h1, a_src1, a_dst1, als1, ald1, Nn);

  // init layer-1 segment state
  k_fill_i32<<<blk((long long)Nn * NH, T), T, 0, stream>>>(m1, ORD_NEG_INF, Nn * NH);
  k_fill_f32<<<blk((long long)Nn * NH, T), T, 0, stream>>>(s1, 0.f, Nn * NH);
  k_fill_f32<<<blk((long long)Nn * 256, T), T, 0, stream>>>(agg1, 0.f, Nn * 256);

  // layer-1 edge softmax + aggregate
  k_edge_max1<<<blk(Etot, T), T, 0, stream>>>(ei, E, Etot, als1, ald1, m1);
  k_edge_sum1<<<blk(Etot, T), T, 0, stream>>>(ei, E, Etot, als1, ald1, m1, s1, ex1);
  k_edge_agg1<<<Etot, 256, 0, stream>>>(ei, E, Etot, h1, ex1, s1, agg1);

  // bias + ELU + convert to f16
  k_elu1<<<blk((long long)Nn * 256, T), T, 0, stream>>>(agg1, b1, h2h, Nn * 256);

  // GEMM2 (WMMA): g = h2 @ W2
  k_gemm2<<<blk((long long)(Nn / 16) * 32, T), T, 0, stream>>>(h2h, W2T, g, Nn);

  // layer-2 attention logits
  k_alpha2<<<blk(Nn, T), T, 0, stream>>>(g, a_src2, a_dst2, als2, ald2, Nn);

  // init layer-2 segment state
  k_fill_i32<<<blk(Nn, T), T, 0, stream>>>(m2, ORD_NEG_INF, Nn);
  k_fill_f32<<<blk(Nn, T), T, 0, stream>>>(s2, 0.f, Nn);
  k_fill_f32<<<blk((long long)Nn * NC, T), T, 0, stream>>>(agg2, 0.f, Nn * NC);

  // layer-2 edge softmax + aggregate
  k_edge_max2<<<blk(Etot, T), T, 0, stream>>>(ei, E, Etot, als2, ald2, m2);
  k_edge_sum2<<<blk(Etot, T), T, 0, stream>>>(ei, E, Etot, als2, ald2, m2, s2, ex2);
  k_edge_agg2<<<blk((long long)Etot * NC, T), T, 0, stream>>>(ei, E, Etot, g, ex2, s2, agg2);

  // final bias + log_softmax
  k_logsoftmax<<<blk(Nn, T), T, 0, stream>>>(agg2, b2, out, Nn);
}
